// BatchGridSubsampling_3487513444655
// MI455X (gfx1250) — matrix operand order repaired and checked by reading, not compiled, not run on hardware
//
#include <hip/hip_runtime.h>
#include <hip/hip_bf16.h>

// ---------- problem constants (from reference) ----------
#define NPTS   (1u << 20)      // 16 * 65536 points
#define BCL    16              // clouds
#define CFEAT  64              // feature channels
#define DLVOX  0.04f           // voxel edge
#define CPC    32768           // 32x32x32 dense cells per cloud (coords <= 24)
#define NCELLS (BCL * CPC)     // 524288 = 2^19
#define SCAN_ITEMS 1024        // cells per scan block
#define NSCANBLK  (NCELLS / SCAN_ITEMS)  // 512 (32 blocks/cloud -> cloud-aligned)
#define NTILES (NPTS / 32)     // 32768 sorted-point tiles

typedef __attribute__((ext_vector_type(16))) _Float16 v16h;
typedef __attribute__((ext_vector_type(8)))  float    v8f;

// ---------- utility ----------
__global__ void k_fill_u32(unsigned* p, unsigned val, size_t n) {
    size_t i = (size_t)blockIdx.x * blockDim.x + threadIdx.x;
    size_t stride = (size_t)gridDim.x * blockDim.x;
    for (; i < n; i += stride) p[i] = val;
}

__device__ __forceinline__ int find_bid(const int* __restrict__ blen, int i) {
    int b = 0;
    while (b < BCL - 1 && i >= blen[b + 1]) b++;
    return b;
}

// ---------- per-cloud voxel-grid origin ----------
__global__ void k_vmin(const float* __restrict__ pts, const int* __restrict__ blen,
                       int* __restrict__ vmin) {
    int i = blockIdx.x * blockDim.x + threadIdx.x;
    if (i >= (int)NPTS) return;
    int b = find_bid(blen, i);
    for (int d = 0; d < 3; d++) {
        int v = (int)floorf(pts[(size_t)i * 3 + d] / DLVOX);
        atomicMin(&vmin[b * 3 + d], v);
    }
}

// ---------- cell id per point + histogram ----------
__global__ void k_cell(const float* __restrict__ pts, const int* __restrict__ blen,
                       const int* __restrict__ vmin, int* __restrict__ cellOf,
                       int* __restrict__ counts) {
    int i = blockIdx.x * blockDim.x + threadIdx.x;
    if (i >= (int)NPTS) return;
    int b = find_bid(blen, i);
    int vx = (int)floorf(pts[(size_t)i * 3 + 0] / DLVOX) - vmin[b * 3 + 0];
    int vy = (int)floorf(pts[(size_t)i * 3 + 1] / DLVOX) - vmin[b * 3 + 1];
    int vz = (int)floorf(pts[(size_t)i * 3 + 2] / DLVOX) - vmin[b * 3 + 2];
    // lexicographic (b,vx,vy,vz) == ascending packed-key order of the reference
    int cell = (b << 15) | (vx << 10) | (vy << 5) | vz;
    cellOf[i] = cell;
    atomicAdd(&counts[cell], 1);
}

// ---------- 3-phase exclusive scan over NCELLS (mode 0: counts, mode 1: occupied flag) ----
__global__ void k_scan_blocksum(const int* __restrict__ src, int mode,
                                int* __restrict__ blockSums) {
    __shared__ int sm[256];
    int base = blockIdx.x * SCAN_ITEMS + threadIdx.x * 4;
    int s = 0;
    for (int j = 0; j < 4; j++) { int v = src[base + j]; s += mode ? (v > 0) : v; }
    sm[threadIdx.x] = s;
    __syncthreads();
    for (int o = 128; o > 0; o >>= 1) {
        if ((int)threadIdx.x < o) sm[threadIdx.x] += sm[threadIdx.x + o];
        __syncthreads();
    }
    if (threadIdx.x == 0) blockSums[blockIdx.x] = sm[0];
}

__global__ void k_scan_top(int* __restrict__ blockSums, int* __restrict__ poolBatch) {
    __shared__ int sm[NSCANBLK];
    int t = threadIdx.x;
    int v = blockSums[t];
    sm[t] = v;
    __syncthreads();
    for (int o = 1; o < NSCANBLK; o <<= 1) {
        int x = (t >= o) ? sm[t - o] : 0;
        __syncthreads();
        sm[t] += x;
        __syncthreads();
    }
    int excl = sm[t] - v;
    blockSums[t] = excl;
    if (poolBatch) {
        if ((t & 31) == 0) poolBatch[t >> 5] = excl;     // 32 scan-blocks per cloud
        if (t == NSCANBLK - 1) poolBatch[BCL] = sm[t];   // total occupied voxels
    }
}

__global__ void k_scan_out(const int* __restrict__ src, int mode,
                           const int* __restrict__ blockSums, int* __restrict__ dst) {
    __shared__ int sm[256];
    int base = blockIdx.x * SCAN_ITEMS + threadIdx.x * 4;
    int vals[4]; int s = 0;
    for (int j = 0; j < 4; j++) { int v = src[base + j]; vals[j] = mode ? (v > 0) : v; s += vals[j]; }
    sm[threadIdx.x] = s;
    __syncthreads();
    for (int o = 1; o < 256; o <<= 1) {
        int x = ((int)threadIdx.x >= o) ? sm[threadIdx.x - o] : 0;
        __syncthreads();
        sm[threadIdx.x] += x;
        __syncthreads();
    }
    int run = blockSums[blockIdx.x] + (sm[threadIdx.x] - s);
    for (int j = 0; j < 4; j++) { dst[base + j] = run; run += vals[j]; }
}

// ---------- counting-sort scatter ----------
__global__ void k_scatter(const int* __restrict__ cellOf, const int* __restrict__ pointStart,
                          const int* __restrict__ rowIndex, int* __restrict__ cursor,
                          int* __restrict__ sortedIdx, int* __restrict__ rowSorted) {
    int i = blockIdx.x * blockDim.x + threadIdx.x;
    if (i >= (int)NPTS) return;
    int cell = cellOf[i];
    int slot = atomicAdd(&cursor[cell], 1);
    int s = pointStart[cell] + slot;
    sortedIdx[s] = i;
    rowSorted[s] = rowIndex[cell];
}

// ---------- WMMA segmented accumulate: one wave per 32 sorted points ----------
// Features staged via CDNA5 async global->LDS DMA (ASYNCcnt), f32 in LDS,
// converted to f16 only when building the B operand. Row stride 68 floats
// (272B, 16B aligned for B128 destinations).
#define WAVES_ACC 4
#define TPB_ACC   (WAVES_ACC * 32)
#define FSTRIDE   68
__global__ void __launch_bounds__(TPB_ACC)
k_accum(const float* __restrict__ pts, const float* __restrict__ feats,
        const int* __restrict__ sortedIdx, const int* __restrict__ rowSorted,
        float* __restrict__ poolPts, float* __restrict__ poolFeats) {
    __shared__ float smf[WAVES_ACC * 32 * FSTRIDE];   // ~34.8 KB, LDS offset 0
    int lane = threadIdx.x & 31;
    int w    = threadIdx.x >> 5;
    int tile = blockIdx.x * WAVES_ACC + w;
    int s    = tile * 32 + lane;

    int idx = sortedIdx[s];
    int r   = rowSorted[s];
    int r0    = __builtin_amdgcn_readfirstlane(r);                  // lane 0's row
    int rLast = __builtin_amdgcn_readfirstlane(__shfl(r, 31, 32));  // lane 31's row

    // --- launch async DMA: this lane's point feature row (256B) -> LDS ---
    // Pass the LDS row pointer into the asm so `smf` escapes: the compiler must
    // then treat the staging region as written by the asm (it has no visible
    // stores otherwise, and would fold the later B-operand loads to undef).
    float* myrow = &smf[(w * 32 + lane) * FSTRIDE];
    unsigned ldsrow = (unsigned)((w * 32 + lane) * FSTRIDE) * 4u;
    unsigned long long g = (unsigned long long)(const void*)(feats + (size_t)idx * CFEAT);
#pragma unroll
    for (int c = 0; c < 16; c++) {
        unsigned la = ldsrow + (unsigned)(c * 16);
        unsigned long long ga = g + (unsigned long long)(c * 16);
        asm volatile("global_load_async_to_lds_b128 %0, %1, off"
                     :: "v"(la), "v"(ga), "v"(myrow) : "memory");
    }

    // --- overlap with DMA: xyz sums, boundary-sharing flags, indicator build ---
    for (int d = 0; d < 3; d++)
        atomicAdd(&poolPts[(size_t)r * 3 + d], pts[(size_t)idx * 3 + d]);

    int prevR = (tile > 0) ? rowSorted[tile * 32 - 1] : -1;
    int nextR = (tile < NTILES - 1) ? rowSorted[tile * 32 + 32] : -1;
    bool shLo = (__builtin_amdgcn_readfirstlane(prevR) == r0);    // row r0 continues left
    bool shHi = (__builtin_amdgcn_readfirstlane(nextR) == rLast); // row rLast continues right

    int hi8 = (lane >> 4) << 3;   // 0 or 8
    int m   = lane & 15;
    int rk[16];
#pragma unroll
    for (int j = 0; j < 16; j++) {
        int k = (hi8 << 1) + j;  // A-matrix 16x32 f16 K mapping (contiguous: k = 16h + j, matches B)
        rk[j] = __shfl(r, k, 32);                // ds_bpermute
    }
    v16h Alo, Ahi;
#pragma unroll
    for (int j = 0; j < 16; j++) {
        Alo[j] = (_Float16)((rk[j] == r0 + m)      ? 1.0f : 0.0f);
        Ahi[j] = (_Float16)((rk[j] == r0 + 16 + m) ? 1.0f : 0.0f);
    }
    bool doHi = (rLast - r0) >= 16;  // scalar -> uniform branch, EXEC all-ones at WMMA

    // Wait for DMA; the escaped pointer + memory clobber pins the B-operand
    // LDS loads below this point.
    const float* wb = &smf[w * 32 * FSTRIDE];
    asm volatile("s_wait_asynccnt 0" :: "v"(wb) : "memory");

    int pb = (lane >> 4) << 4;  // 0 or 16: which 16 sorted points this lane's B half covers
#pragma unroll
    for (int nb = 0; nb < 4; nb++) {
        v16h Bv;
#pragma unroll
        for (int j = 0; j < 16; j++)
            Bv[j] = (_Float16)wb[(pb + j) * FSTRIDE + nb * 16 + m];
        v8f c0 = {};
        v8f dlo = __builtin_amdgcn_wmma_f32_16x16x32_f16(false, Alo, false, Bv,
                                                         (short)0, c0, false, false);
#pragma unroll
        for (int gg = 0; gg < 8; gg++) {
            int ro = r0 + gg + hi8;
            if (ro <= rLast) {
                float* dst = &poolFeats[(size_t)ro * CFEAT + nb * 16 + m];
                // boundary rows may be split across tiles -> RMW; interior rows are
                // complete in this tile -> plain store (no atomic)
                if ((ro == r0 && shLo) || (ro == rLast && shHi)) atomicAdd(dst, dlo[gg]);
                else                                             *dst = dlo[gg];
            }
        }
        if (doHi) {
            v8f dhi = __builtin_amdgcn_wmma_f32_16x16x32_f16(false, Ahi, false, Bv,
                                                             (short)0, c0, false, false);
#pragma unroll
            for (int gg = 0; gg < 8; gg++) {
                int ro = r0 + 16 + gg + hi8;
                if (ro <= rLast) {
                    float* dst = &poolFeats[(size_t)ro * CFEAT + nb * 16 + m];
                    if (ro == rLast && shHi) atomicAdd(dst, dhi[gg]);
                    else                     *dst = dhi[gg];
                }
            }
        }
    }
}

// ---------- sums -> means ----------
__global__ void k_normalize(const int* __restrict__ counts, const int* __restrict__ rowIndex,
                            float* __restrict__ poolPts, float* __restrict__ poolFeats) {
    int i = blockIdx.x * blockDim.x + threadIdx.x;
    if (i >= (int)NCELLS) return;
    int c = counts[i];
    if (c <= 0) return;
    int row = rowIndex[i];
    float inv = 1.0f / (float)c;
    for (int d = 0; d < 3; d++)     poolPts[(size_t)row * 3 + d] *= inv;
    for (int f = 0; f < CFEAT; f++) poolFeats[(size_t)row * CFEAT + f] *= inv;
}

extern "C" void kernel_launch(void* const* d_in, const int* in_sizes, int n_in,
                              void* d_out, int out_size, void* d_ws, size_t ws_size,
                              hipStream_t stream) {
    const float* pts  = (const float*)d_in[0];
    const float* fts  = (const float*)d_in[1];
    const int*   blen = (const int*)d_in[2];

    float* out       = (float*)d_out;
    float* poolPts   = out;
    float* poolFeats = out + (size_t)3 * NPTS;
    int*   poolBatch = (int*)(out + (size_t)(3 + CFEAT) * NPTS);

    int* ws         = (int*)d_ws;
    int* vmin       = ws;                    // 64 ints
    int* counts     = ws + 64;               // NCELLS
    int* pointStart = counts + NCELLS;       // NCELLS
    int* rowIndex   = pointStart + NCELLS;   // NCELLS
    int* cursor     = rowIndex + NCELLS;     // NCELLS
    int* cellOf     = cursor + NCELLS;       // NPTS
    int* sortedIdx  = cellOf + NPTS;         // NPTS
    int* rowSorted  = sortedIdx + NPTS;      // NPTS
    int* blockSums  = rowSorted + NPTS;      // NSCANBLK

    // init: zero output (padded rows must be zero), histogram, cursors; vmin = +inf
    k_fill_u32<<<4096, 256, 0, stream>>>((unsigned*)d_out, 0u, (size_t)out_size);
    k_fill_u32<<<512, 256, 0, stream>>>((unsigned*)counts, 0u, (size_t)NCELLS);
    k_fill_u32<<<512, 256, 0, stream>>>((unsigned*)cursor, 0u, (size_t)NCELLS);
    k_fill_u32<<<1, 64, 0, stream>>>((unsigned*)vmin, 0x7FFFFFFFu, 64);

    k_vmin<<<NPTS / 256, 256, 0, stream>>>(pts, blen, vmin);
    k_cell<<<NPTS / 256, 256, 0, stream>>>(pts, blen, vmin, cellOf, counts);

    // scan #1: point offsets per cell
    k_scan_blocksum<<<NSCANBLK, 256, 0, stream>>>(counts, 0, blockSums);
    k_scan_top<<<1, NSCANBLK, 0, stream>>>(blockSums, nullptr);
    k_scan_out<<<NSCANBLK, 256, 0, stream>>>(counts, 0, blockSums, pointStart);

    // scan #2: compacted row index per occupied cell (+ pool_batch offsets)
    k_scan_blocksum<<<NSCANBLK, 256, 0, stream>>>(counts, 1, blockSums);
    k_scan_top<<<1, NSCANBLK, 0, stream>>>(blockSums, poolBatch);
    k_scan_out<<<NSCANBLK, 256, 0, stream>>>(counts, 1, blockSums, rowIndex);

    k_scatter<<<NPTS / 256, 256, 0, stream>>>(cellOf, pointStart, rowIndex, cursor,
                                              sortedIdx, rowSorted);
    k_accum<<<NTILES / WAVES_ACC, TPB_ACC, 0, stream>>>(pts, fts, sortedIdx, rowSorted,
                                                        poolPts, poolFeats);
    k_normalize<<<NCELLS / 256, 256, 0, stream>>>(counts, rowIndex, poolPts, poolFeats);
}
